// BrainNetv2__80418967650908
// MI455X (gfx1250) — compile-verified, matrix operand order checked by reading, not run on hardware
//
#include <hip/hip_runtime.h>
#include <hip/hip_bf16.h>

// ---------------------------------------------------------------------------
// BrainNetv2 forward for MI455X (gfx1250, wave32, WMMA 16x16x32 f16).
// - All matmuls on v_wmma_f32_16x16x32_f16 (f32 accumulate).
// - Instance norms fused onto the WMMA accumulator layout (half-wave shuffles).
// - All WMMA fragments load as two 128-bit transactions per lane.
// - B operands staged in LDS once per block (removes 8x per-wave L2 streaming
//   and replaces long-latency global loads in the inner loop with ds_load).
// ---------------------------------------------------------------------------

typedef __attribute__((ext_vector_type(16))) _Float16 v16h;
typedef __attribute__((ext_vector_type(8)))  _Float16 v8h;
typedef __attribute__((ext_vector_type(8)))  float    v8f;

#define B_    1024
#define C_    64
#define T_    1024
#define H_    128
#define DM_   256
#define HEAD_ 8
#define DK_   32

__device__ __forceinline__ v8f wmma_f16(v16h a, v16h b, v8f c) {
  // (neg_a, A, neg_b, B, c_mod, C, reuse_a, reuse_b)
  return __builtin_amdgcn_wmma_f32_16x16x32_f16(false, a, false, b, (short)0, c,
                                                false, false);
}

// A fragment (16x32 f16): element A[m][k] at base[m*ld + k].
// ISA layout: lane l -> row (l&15); VGPR i -> K = (i>>2)*16 + (l>>4)*8 + (i&3)*2.
// Per lane: two contiguous 16-byte chunks at [half*8] and [16 + half*8].
// Requires base and ld to be multiples of 8 elements (true everywhere here).
// Works for global or LDS pointers (address space inferred after inlining).
__device__ __forceinline__ v16h load_frag(const _Float16* base, int ld) {
  const int lane = threadIdx.x & 31;
  const int m = lane & 15;
  const int half = lane >> 4;
  const _Float16* p = base + m * ld + half * 8;
  const v8h lo = *(const v8h*)(p);
  const v8h hi = *(const v8h*)(p + 16);
  v16h a;
#pragma unroll
  for (int i = 0; i < 8; ++i) { a[i] = lo[i]; a[8 + i] = hi[i]; }
  return a;
}
// B fragment (32x16 f16) supplied transposed: element B[k][n] at base[n*ld + k].
#define load_fragA  load_frag
#define load_fragBT load_frag

__device__ __forceinline__ float elu_f(float v) {
  return v > 0.f ? v : (__expf(v) - 1.f);
}

// Per-row instance norm + ELU over a 16-row x 256-col strip held in 16 WMMA
// accumulator tiles, then store as f16. Row stats need only a 16-lane shuffle
// reduction (lanes 0-15 and 16-31 hold disjoint row halves in the C layout).
__device__ __forceinline__ void rownorm_elu_store(v8f (&fragC)[16],
                                                  _Float16* outbase, int m0) {
  const int lane = threadIdx.x & 31;
  const int half = lane >> 4;
  const int col0 = lane & 15;
#pragma unroll
  for (int i = 0; i < 8; ++i) {
    float s = 0.f, ss = 0.f;
#pragma unroll
    for (int nt = 0; nt < 16; ++nt) { float v = fragC[nt][i]; s += v; ss += v * v; }
#pragma unroll
    for (int m = 1; m < 16; m <<= 1) {
      s  += __shfl_xor(s, m, 32);
      ss += __shfl_xor(ss, m, 32);
    }
    const float mean = s * (1.f / 256.f);
    const float var  = ss * (1.f / 256.f) - mean * mean;
    const float inv  = rsqrtf(var + 1e-5f);
    const int row = m0 + i + 8 * half;
    _Float16* orow = outbase + (size_t)row * DM_;
#pragma unroll
    for (int nt = 0; nt < 16; ++nt) {
      float v = (fragC[nt][i] - mean) * inv;
      orow[nt * 16 + col0] = (_Float16)elu_f(v);
    }
  }
}

// ---------------------------------------------------------------------------
// K0: f32 -> f16 weight conversion
// ---------------------------------------------------------------------------
__global__ __launch_bounds__(256) void cvt_f16_kernel(const float* __restrict__ src,
                                                      _Float16* __restrict__ dst,
                                                      int n) {
  int i = blockIdx.x * 256 + threadIdx.x;
  if (i < n) dst[i] = (_Float16)src[i];
}

// ---------------------------------------------------------------------------
// K1: block1 = scalar affine -> instance norm over T -> ELU -> avgpool(4).
// One block per (b,c) row; thread t owns pool bin t. Output TRANSPOSED as
// S1t[b][t][c] so conv2's B operand has contiguous K (=c) per lane.
// ---------------------------------------------------------------------------
__global__ __launch_bounds__(256) void block1_kernel(const float* __restrict__ x,
                                                     const float* __restrict__ w1,
                                                     const float* __restrict__ b1,
                                                     _Float16* __restrict__ S1t) {
  const int row = blockIdx.x;                 // b*C + c
  const int b = row >> 6, c = row & 63;
  const float w = w1[0], bb = b1[0];
  const float4* xr = (const float4*)(x + (size_t)row * T_);
  const int t = threadIdx.x;

  const float4 xv = xr[t];
  float z[4] = {xv.x * w + bb, xv.y * w + bb, xv.z * w + bb, xv.w * w + bb};
  float s = 0.f, ss = 0.f;
#pragma unroll
  for (int j = 0; j < 4; ++j) { s += z[j]; ss += z[j] * z[j]; }
#pragma unroll
  for (int m = 1; m < 32; m <<= 1) {
    s  += __shfl_xor(s, m, 32);
    ss += __shfl_xor(ss, m, 32);
  }
  __shared__ float rs[8], rss[8];
  const int wid = t >> 5, ln = t & 31;
  if (ln == 0) { rs[wid] = s; rss[wid] = ss; }
  __syncthreads();
  float S = 0.f, SS = 0.f;
#pragma unroll
  for (int i = 0; i < 8; ++i) { S += rs[i]; SS += rss[i]; }
  const float mean = S * (1.f / 1024.f);
  const float var  = SS * (1.f / 1024.f) - mean * mean;
  const float inv  = rsqrtf(var + 1e-5f);
  float p = 0.f;
#pragma unroll
  for (int j = 0; j < 4; ++j) p += elu_f((z[j] - mean) * inv);
  S1t[(size_t)b * (DM_ * C_) + t * C_ + c] = (_Float16)(p * 0.25f);
}

// ---------------------------------------------------------------------------
// K2: conv2 (128x64 @ 64x256 per batch) + instance norm over d + ELU.
// Per-batch activation (32 KB) staged in LDS once per block; each of the 8
// waves then reads B fragments from LDS. conv2_b omitted (cancelled by norm).
// ---------------------------------------------------------------------------
__global__ __launch_bounds__(256) void conv2_kernel(const _Float16* __restrict__ S1t,
                                                    const _Float16* __restrict__ W2h,
                                                    _Float16* __restrict__ Z2n) {
  const int b = blockIdx.x;
  const int m0 = (threadIdx.x >> 5) * 16;
  __shared__ _Float16 sB[DM_ * C_];           // 32 KB, layout (t, c)
  {
    const v8h* src = (const v8h*)(S1t + (size_t)b * (DM_ * C_));
    v8h* dst = (v8h*)sB;
#pragma unroll
    for (int it = 0; it < (DM_ * C_ / 8) / 256; ++it)
      dst[it * 256 + threadIdx.x] = src[it * 256 + threadIdx.x];
  }
  __syncthreads();

  v8f zero = {};
  v8f fragC[16];
#pragma unroll
  for (int nt = 0; nt < 16; ++nt) fragC[nt] = zero;
#pragma unroll
  for (int ks = 0; ks < 2; ++ks) {
    v16h a = load_fragA(W2h + m0 * C_ + ks * 32, C_);
#pragma unroll
    for (int nt = 0; nt < 16; ++nt) {
      v16h bf = load_fragBT(sB + (nt * 16) * C_ + ks * 32, C_);
      fragC[nt] = wmma_f16(a, bf, fragC[nt]);
    }
  }
  rownorm_elu_store(fragC, Z2n + (size_t)b * H_ * DM_, m0);
}

// ---------------------------------------------------------------------------
// Shared GEMM body for K3/K4: per-batch (128x256) = A(128x256) @ W^T(256x256),
// W staged in LDS in two 64 KB phases over the N dimension. Accumulators for
// all 16 n-tiles stay live across phases (only the staged half of W changes).
// ---------------------------------------------------------------------------
__device__ __forceinline__ void gemm_256(const _Float16* __restrict__ Ab,
                                         const _Float16* __restrict__ W,
                                         _Float16* __restrict__ sW,  // 128*256
                                         int m0, v8f (&fragC)[16]) {
#pragma unroll
  for (int p = 0; p < 2; ++p) {
    __syncthreads();                          // previous phase readers done
    {
      const v8h* src = (const v8h*)(W + (size_t)p * 128 * DM_);
      v8h* dst = (v8h*)sW;
#pragma unroll
      for (int it = 0; it < (128 * DM_ / 8) / 256; ++it)
        dst[it * 256 + threadIdx.x] = src[it * 256 + threadIdx.x];
    }
    __syncthreads();
    for (int ks = 0; ks < 8; ++ks) {          // K = 256
      v16h a = load_fragA(Ab + m0 * DM_ + ks * 32, DM_);
#pragma unroll
      for (int nt = 0; nt < 8; ++nt) {
        v16h bf = load_fragBT(sW + (nt * 16) * DM_ + ks * 32, DM_);
        fragC[p * 8 + nt] = wmma_f16(a, bf, fragC[p * 8 + nt]);
      }
    }
  }
}

// ---------------------------------------------------------------------------
// K3: time-attention gate GEMM (z @ ta_w^T + ta_b), elementwise z*gate,
//     instance norm + ELU.
// ---------------------------------------------------------------------------
__global__ __launch_bounds__(256) void tatt_kernel(const _Float16* __restrict__ Z2n,
                                                   const _Float16* __restrict__ TAWh,
                                                   const float* __restrict__ ta_b,
                                                   _Float16* __restrict__ Zg) {
  const int b = blockIdx.x;
  const int m0 = (threadIdx.x >> 5) * 16;
  const _Float16* Ab = Z2n + (size_t)b * H_ * DM_;
  __shared__ _Float16 sW[128 * DM_];          // 64 KB
  v8f zero = {};
  v8f fragC[16];
#pragma unroll
  for (int nt = 0; nt < 16; ++nt) fragC[nt] = zero;
  gemm_256(Ab, TAWh, sW, m0, fragC);

  // gate -> z * (gate + ta_b)
  const int lane = threadIdx.x & 31;
  const int half = lane >> 4;
  const int col0 = lane & 15;
#pragma unroll
  for (int i = 0; i < 8; ++i) {
    const int row = m0 + i + 8 * half;
    const _Float16* zr = Ab + (size_t)row * DM_;
#pragma unroll
    for (int nt = 0; nt < 16; ++nt) {
      const int col = nt * 16 + col0;
      fragC[nt][i] = (float)zr[col] * (fragC[nt][i] + ta_b[col]);
    }
  }
  rownorm_elu_store(fragC, Zg + (size_t)b * H_ * DM_, m0);
}

// ---------------------------------------------------------------------------
// K4: q/k/v projections (no bias, no norm): Zg @ W^T. blockIdx.y selects W.
// Q, K stored row-major (b,h,e); V stored TRANSPOSED (b,e,h) so attention
// phase 2's B operand has contiguous K (=g) per lane.
// ---------------------------------------------------------------------------
__global__ __launch_bounds__(256) void qkv_kernel(const _Float16* __restrict__ Zg,
                                                  const _Float16* __restrict__ WQh,
                                                  const _Float16* __restrict__ WKh,
                                                  const _Float16* __restrict__ WVh,
                                                  _Float16* __restrict__ Q,
                                                  _Float16* __restrict__ K,
                                                  _Float16* __restrict__ Vt) {
  const int b = blockIdx.x;
  const int which = blockIdx.y;
  const _Float16* W = (which == 0) ? WQh : (which == 1) ? WKh : WVh;
  const int m0 = (threadIdx.x >> 5) * 16;
  const _Float16* Ab = Zg + (size_t)b * H_ * DM_;
  __shared__ _Float16 sW[128 * DM_];          // 64 KB
  v8f zero = {};
  v8f fragC[16];
#pragma unroll
  for (int nt = 0; nt < 16; ++nt) fragC[nt] = zero;
  gemm_256(Ab, W, sW, m0, fragC);

  const int lane = threadIdx.x & 31;
  const int half = lane >> 4;
  const int col0 = lane & 15;
  if (which < 2) {
    _Float16* ob = ((which == 0) ? Q : K) + (size_t)b * H_ * DM_;
#pragma unroll
    for (int i = 0; i < 8; ++i) {
      const int row = m0 + i + 8 * half;
#pragma unroll
      for (int nt = 0; nt < 16; ++nt)
        ob[(size_t)row * DM_ + nt * 16 + col0] = (_Float16)fragC[nt][i];
    }
  } else {
    _Float16* ob = Vt + (size_t)b * H_ * DM_;   // layout (e, h): ob[e*128 + h]
#pragma unroll
    for (int i = 0; i < 8; ++i) {
      const int row = m0 + i + 8 * half;
#pragma unroll
      for (int nt = 0; nt < 16; ++nt)
        ob[(size_t)(nt * 16 + col0) * H_ + row] = (_Float16)fragC[nt][i];
    }
  }
}

// ---------------------------------------------------------------------------
// K5: per-(batch, head) multi-head channel attention. One wave per block.
//   S = q_h @ k_h^T (128x128, K=32, unscaled), row softmax,
//   O = P @ v_h (128x32, K=128, P staged through LDS as f16 A-fragments).
// K-fragments (phase 1) and V-fragments (phase 2) are loop-invariant across
// the 8 M-strips, so they are hoisted into register arrays and loaded once.
// ---------------------------------------------------------------------------
__global__ __launch_bounds__(32) void attn_kernel(const _Float16* __restrict__ Q,
                                                  const _Float16* __restrict__ K,
                                                  const _Float16* __restrict__ Vt,
                                                  const _Float16* __restrict__ Zg,
                                                  float* __restrict__ Of) {
  const int b = blockIdx.x;
  const int head = blockIdx.y;
  __shared__ _Float16 sP[H_ * H_];            // 128x128 f16 = 32 KB

  const _Float16* Qb  = Q  + (size_t)b * H_ * DM_ + head * DK_;
  const _Float16* Kb  = K  + (size_t)b * H_ * DM_ + head * DK_;
  const _Float16* Vbt = Vt + (size_t)b * H_ * DM_ + (size_t)(head * DK_) * H_;
  const int lane = threadIdx.x & 31;
  const int half = lane >> 4;
  const int col0 = lane & 15;
  v8f zero = {};

  // Phase 1: scores + softmax -> sP. K-fragments loaded once.
  v16h kf[8];
#pragma unroll
  for (int nt = 0; nt < 8; ++nt) kf[nt] = load_fragBT(Kb + (nt * 16) * DM_, DM_);
  for (int ms = 0; ms < 8; ++ms) {
    v16h a = load_fragA(Qb + (ms * 16) * DM_, DM_);
    v8f acc[8];
#pragma unroll
    for (int nt = 0; nt < 8; ++nt) acc[nt] = wmma_f16(a, kf[nt], zero);
#pragma unroll
    for (int i = 0; i < 8; ++i) {
      float mx = -3.0e38f;
#pragma unroll
      for (int nt = 0; nt < 8; ++nt) mx = fmaxf(mx, acc[nt][i]);
#pragma unroll
      for (int m = 1; m < 16; m <<= 1) mx = fmaxf(mx, __shfl_xor(mx, m, 32));
      float s = 0.f;
#pragma unroll
      for (int nt = 0; nt < 8; ++nt) {
        float e = __expf(acc[nt][i] - mx);
        acc[nt][i] = e; s += e;
      }
#pragma unroll
      for (int m = 1; m < 16; m <<= 1) s += __shfl_xor(s, m, 32);
      const float inv = 1.f / s;
      const int row = ms * 16 + i + 8 * half;
#pragma unroll
      for (int nt = 0; nt < 8; ++nt)
        sP[row * H_ + nt * 16 + col0] = (_Float16)(acc[nt][i] * inv);
    }
  }
  __syncthreads();

  // Phase 2: O = P @ v_h. V-fragments loaded once.
  v16h vf[2][4];
#pragma unroll
  for (int nt = 0; nt < 2; ++nt)
#pragma unroll
    for (int ks = 0; ks < 4; ++ks)
      vf[nt][ks] = load_fragBT(Vbt + (nt * 16) * H_ + ks * 32, H_);

  for (int ms = 0; ms < 8; ++ms) {
    v16h af[4];
#pragma unroll
    for (int ks = 0; ks < 4; ++ks)
      af[ks] = load_fragA(&sP[(ms * 16) * H_ + ks * 32], H_);
#pragma unroll
    for (int nt = 0; nt < 2; ++nt) {
      v8f acc = zero;
#pragma unroll
      for (int ks = 0; ks < 4; ++ks) acc = wmma_f16(af[ks], vf[nt][ks], acc);
#pragma unroll
      for (int i = 0; i < 8; ++i) {
        const int row = ms * 16 + i + 8 * half;
        const int c = head * DK_ + nt * 16 + col0;
        const size_t idx = (size_t)b * H_ * DM_ + (size_t)row * DM_ + c;
        Of[idx] = elu_f(acc[i]) + (float)Zg[idx];
      }
    }
  }
}

// ---------------------------------------------------------------------------
// K6: final instance norm per (b,h) row + classifier + 2-way softmax.
// One block per batch, 256 threads (= one row of d_m per iteration).
// ---------------------------------------------------------------------------
__global__ __launch_bounds__(256) void final_kernel(const float* __restrict__ Of,
                                                    const float* __restrict__ cls_w,
                                                    const float* __restrict__ cls_b,
                                                    float* __restrict__ out) {
  const int b = blockIdx.x;
  const int t = threadIdx.x;
  const int wid = t >> 5, ln = t & 31;
  __shared__ float red[16];
  float a0 = 0.f, a1 = 0.f;
  for (int h = 0; h < H_; ++h) {
    const float v = Of[(size_t)b * H_ * DM_ + (size_t)h * DM_ + t];
    float s = v, ss = v * v;
#pragma unroll
    for (int m = 1; m < 32; m <<= 1) {
      s  += __shfl_xor(s, m, 32);
      ss += __shfl_xor(ss, m, 32);
    }
    if (ln == 0) { red[wid] = s; red[8 + wid] = ss; }
    __syncthreads();
    float S = 0.f, SS = 0.f;
#pragma unroll
    for (int i = 0; i < 8; ++i) { S += red[i]; SS += red[8 + i]; }
    __syncthreads();
    const float mean = S * (1.f / 256.f);
    const float var  = SS * (1.f / 256.f) - mean * mean;
    const float zn = (v - mean) * rsqrtf(var + 1e-5f);
    a0 += zn * cls_w[h * DM_ + t];
    a1 += zn * cls_w[H_ * DM_ + h * DM_ + t];
  }
#pragma unroll
  for (int m = 1; m < 32; m <<= 1) {
    a0 += __shfl_xor(a0, m, 32);
    a1 += __shfl_xor(a1, m, 32);
  }
  if (ln == 0) { red[wid] = a0; red[8 + wid] = a1; }
  __syncthreads();
  if (t == 0) {
    float l0 = cls_b[0], l1 = cls_b[1];
#pragma unroll
    for (int i = 0; i < 8; ++i) { l0 += red[i]; l1 += red[8 + i]; }
    const float mx = fmaxf(l0, l1);
    const float e0 = __expf(l0 - mx), e1 = __expf(l1 - mx);
    const float inv = 1.f / (e0 + e1);
    out[b * 2 + 0] = e0 * inv;
    out[b * 2 + 1] = e1 * inv;
  }
}

// ---------------------------------------------------------------------------
extern "C" void kernel_launch(void* const* d_in, const int* in_sizes, int n_in,
                              void* d_out, int out_size, void* d_ws, size_t ws_size,
                              hipStream_t stream) {
  (void)in_sizes; (void)n_in; (void)out_size; (void)ws_size;
  const float* x       = (const float*)d_in[0];
  const float* conv1_w = (const float*)d_in[1];
  const float* conv1_b = (const float*)d_in[2];
  const float* conv2_w = (const float*)d_in[3];
  // d_in[4] (conv2_b): per-row bias cancelled exactly by the following inorm.
  const float* ta_w    = (const float*)d_in[5];
  const float* ta_b    = (const float*)d_in[6];
  const float* wq      = (const float*)d_in[7];
  const float* wk      = (const float*)d_in[8];
  const float* wv      = (const float*)d_in[9];
  const float* cls_w   = (const float*)d_in[10];
  const float* cls_b   = (const float*)d_in[11];
  float* out = (float*)d_out;

  char* ws = (char*)d_ws;
  size_t off = 0;
  auto alloc = [&](size_t bytes) -> void* {
    void* p = ws + off;
    off = (off + bytes + 255) & ~(size_t)255;
    return p;
  };
  _Float16* W2h  = (_Float16*)alloc((size_t)H_ * C_ * 2);
  _Float16* TAWh = (_Float16*)alloc((size_t)DM_ * DM_ * 2);
  _Float16* WQh  = (_Float16*)alloc((size_t)DM_ * DM_ * 2);
  _Float16* WKh  = (_Float16*)alloc((size_t)DM_ * DM_ * 2);
  _Float16* WVh  = (_Float16*)alloc((size_t)DM_ * DM_ * 2);
  _Float16* S1t  = (_Float16*)alloc((size_t)B_ * C_ * DM_ * 2);   //  32 MB
  _Float16* Z2n  = (_Float16*)alloc((size_t)B_ * H_ * DM_ * 2);   //  64 MB
  _Float16* Zg   = (_Float16*)alloc((size_t)B_ * H_ * DM_ * 2);   //  64 MB
  _Float16* Qb   = (_Float16*)alloc((size_t)B_ * H_ * DM_ * 2);   //  64 MB
  _Float16* Kb   = (_Float16*)alloc((size_t)B_ * H_ * DM_ * 2);   //  64 MB
  _Float16* Vbt  = (_Float16*)alloc((size_t)B_ * H_ * DM_ * 2);   //  64 MB
  float*    Of   = (float*)   alloc((size_t)B_ * H_ * DM_ * 4);   // 128 MB

  // K0: weight conversions
  cvt_f16_kernel<<<(H_ * C_ + 255) / 256, 256, 0, stream>>>(conv2_w, W2h, H_ * C_);
  cvt_f16_kernel<<<(DM_ * DM_ + 255) / 256, 256, 0, stream>>>(ta_w, TAWh, DM_ * DM_);
  cvt_f16_kernel<<<(DM_ * DM_ + 255) / 256, 256, 0, stream>>>(wq, WQh, DM_ * DM_);
  cvt_f16_kernel<<<(DM_ * DM_ + 255) / 256, 256, 0, stream>>>(wk, WKh, DM_ * DM_);
  cvt_f16_kernel<<<(DM_ * DM_ + 255) / 256, 256, 0, stream>>>(wv, WVh, DM_ * DM_);

  // K1: block1 (affine + inorm + elu + avgpool), transposed output
  block1_kernel<<<B_ * C_, 256, 0, stream>>>(x, conv1_w, conv1_b, S1t);

  // K2: conv2 GEMM + inorm + elu (B staged in LDS)
  conv2_kernel<<<B_, 256, 0, stream>>>(S1t, W2h, Z2n);

  // K3: time-attention gate + inorm + elu (W staged in LDS, 2 phases)
  tatt_kernel<<<B_, 256, 0, stream>>>(Z2n, TAWh, ta_b, Zg);

  // K4: q/k/v projections (V transposed; W staged in LDS, 2 phases)
  qkv_kernel<<<dim3(B_, 3), 256, 0, stream>>>(Zg, WQh, WKh, WVh, Qb, Kb, Vbt);

  // K5: attention per (batch, head)
  attn_kernel<<<dim3(B_, HEAD_), 32, 0, stream>>>(Qb, Kb, Vbt, Zg, Of);

  // K6: final norm + classifier + softmax
  final_kernel<<<B_, 256, 0, stream>>>(Of, cls_w, cls_b, out);
}
